// NeuralODE_48541720379612
// MI455X (gfx1250) — matrix operand order repaired and checked by reading, not compiled, not run on hardware
//
#include <hip/hip_runtime.h>

// Problem constants
#define NB    4096
#define NT    128
#define HD    128
// LDS strides (in halves), padded to avoid bank conflicts (byte-stride/4 mod 64 = 4
// tiles 16B accesses across all 64 banks in 2 optimal passes)
#define SA    168     // activation row stride (128 data + 32 xm/pad + 8 pad)
#define SWG   392     // GRU weightT row stride (384 + 8)
#define SW1   136     // enc_W1T row stride (128 + 8)
#define SW2   264     // enc_W2T row stride (256 + 8)
#define SWO   136     // ode_W*T row stride (128 + 8)
#define SRB   136     // GRU r-gate f16 buffer row stride

typedef _Float16 v16h __attribute__((ext_vector_type(16)));
typedef _Float16 v8h  __attribute__((ext_vector_type(8)));
typedef float    v8f  __attribute__((ext_vector_type(8)));

__device__ __forceinline__ v8f wmma_f16(v16h a, v16h b, v8f c) {
  // D(16x16 f32) = A(16x32 f16) * B(32x16 f16) + C
  return __builtin_amdgcn_wmma_f32_16x16x32_f16(false, a, false, b, (short)0, c,
                                                false, false);
}

// A operand: rows m=0..15, cols k0..k0+31 from row-major f16 LDS tile.
// ISA layout: lanes 0-15 -> m=lane, K={k0..k0+7, k0+16..k0+23};
//             lanes 16-31 -> m=lane-16, K={k0+8..k0+15, k0+24..k0+31}
__device__ __forceinline__ v16h ldsA(const _Float16* __restrict__ act, int stride,
                                     int k0, int lane) {
  int m   = lane & 15;
  int sel = (lane >> 4) & 1;
  const _Float16* p = act + m * stride + k0 + sel * 8;
  v8h lo = *(const v8h*)p;
  v8h hi = *(const v8h*)(p + 16);
  v16h r;
#pragma unroll
  for (int i = 0; i < 8; ++i) { r[i] = lo[i]; r[i + 8] = hi[i]; }
  return r;
}

// B operand: rows k0..k0+31, cols n0..n0+15 of W^T (stored [K][N] row-major f16).
// ISA layout: lanes 0-15 -> n=lane, K={k0..k0+7, k0+16..k0+23};
//             lanes 16-31 -> n=lane-16, K={k0+8..k0+15, k0+24..k0+31}
__device__ __forceinline__ v16h ldsB(const _Float16* __restrict__ wt, int stride,
                                     int k0, int n0, int lane) {
  int n   = lane & 15;
  int sel = (lane >> 4) & 1;
  const _Float16* p = wt + (k0 + sel * 8) * stride + n0 + n;
  v16h r;
#pragma unroll
  for (int i = 0; i < 8; ++i) { r[i] = p[i * stride]; r[i + 8] = p[(i + 16) * stride]; }
  return r;
}

// Scatter a D tile (f32) to f16 LDS activation tile.
// D layout: VGPR r, lanes 0-15 -> (m=r, n=lane); lanes 16-31 -> (m=r+8, n=lane-16)
__device__ __forceinline__ void stD_tile(_Float16* __restrict__ act, int stride,
                                         int n0, v8f d, int lane) {
  int n  = lane & 15;
  int mb = (lane >> 4) << 3;
#pragma unroll
  for (int r = 0; r < 8; ++r)
    act[(mb + r) * stride + n0 + n] = (_Float16)d[r];
}

// ---- fast transcendentals (keep the GRU/SELU path on TRANS ops, no IEEE div) ----
__device__ __forceinline__ float fast_rcp(float x) {
#if __has_builtin(__builtin_amdgcn_rcpf)
  return __builtin_amdgcn_rcpf(x);
#else
  return 1.f / x;
#endif
}
__device__ __forceinline__ float sigmoid_f(float x) {
  return fast_rcp(1.f + __expf(-x));
}
__device__ __forceinline__ float fast_tanh(float x) {
#if __has_builtin(__builtin_amdgcn_tanhf)
  return __builtin_amdgcn_tanhf(x);
#else
  float e = __expf(2.f * x);
  return 1.f - 2.f * fast_rcp(e + 1.f);
#endif
}
__device__ __forceinline__ float selu_f(float x) {
  // lambda = 1.0507009873554805, lambda*alpha = 1.7580993408473766
  return x > 0.f ? 1.0507009873554805f * x
                 : 1.7580993408473766f * (__expf(x) - 1.f);
}

// Stage W (global f32, [O][I] row-major) -> LDS f16 transposed Wt[I][stride]
__device__ __forceinline__ void stageWT(const float* __restrict__ W,
                                        _Float16* __restrict__ wt,
                                        int O, int I, int stride, int tid, int nth) {
  for (int idx = tid; idx < O * I; idx += nth) {
    int o = idx / I;
    int i = idx - o * I;
    wt[i * stride + o] = (_Float16)W[idx];
  }
}

// One ODE MLP layer: out = selu(in @ Wt + b). A fragments hoisted (reused by all
// 8 N-tiles) so LDS traffic is 4 A-frags + 32 B-frags per layer.
__device__ __forceinline__ void layer_selu(const _Float16* __restrict__ in,
                                           _Float16* __restrict__ out,
                                           const _Float16* __restrict__ Wt,
                                           const float bias[8], int lane) {
  v16h a0 = ldsA(in, SA, 0,  lane);
  v16h a1 = ldsA(in, SA, 32, lane);
  v16h a2 = ldsA(in, SA, 64, lane);
  v16h a3 = ldsA(in, SA, 96, lane);
#pragma unroll
  for (int nt = 0; nt < 8; ++nt) {
    int n0 = nt * 16;
    v8f c;
#pragma unroll
    for (int r = 0; r < 8; ++r) c[r] = bias[nt];
    c = wmma_f16(a0, ldsB(Wt, SWO, 0,  n0, lane), c);
    c = wmma_f16(a1, ldsB(Wt, SWO, 32, n0, lane), c);
    c = wmma_f16(a2, ldsB(Wt, SWO, 64, n0, lane), c);
    c = wmma_f16(a3, ldsB(Wt, SWO, 96, n0, lane), c);
#pragma unroll
    for (int r = 0; r < 8; ++r) c[r] = selu_f(c[r]);
    stD_tile(out, SA, n0, c, lane);
  }
}

extern "C" __global__ __launch_bounds__(64)
void node_kernel(const float* __restrict__ x, const float* __restrict__ meta,
                 const float* __restrict__ eps,
                 const float* __restrict__ gru_Wih, const float* __restrict__ gru_Whh,
                 const float* __restrict__ gru_bih, const float* __restrict__ gru_bhh,
                 const float* __restrict__ enc_W1, const float* __restrict__ enc_b1,
                 const float* __restrict__ enc_W2, const float* __restrict__ enc_b2,
                 const float* __restrict__ ode_W1, const float* __restrict__ ode_b1,
                 const float* __restrict__ ode_W2, const float* __restrict__ ode_b2,
                 const float* __restrict__ ode_W3, const float* __restrict__ ode_b3,
                 const float* __restrict__ ode_W4, const float* __restrict__ ode_b4,
                 const float* __restrict__ fc_W, const float* __restrict__ fc_b,
                 float* __restrict__ out)
{
  extern __shared__ char smem[];
  const int tid  = threadIdx.x;
  const int wave = tid >> 5;
  const int lane = tid & 31;
  const int nth  = blockDim.x;
  const int row0 = blockIdx.x * 32 + wave * 16;   // 16 batch rows per wave
  const int mb   = (lane >> 4) << 3;              // D-layout m base for this lane
  const int nn   = lane & 15;                     // D-layout n for this lane

  // LDS: [weights 139264B (GRU rbuf16 lives in its tail)][act 2 waves x 2 x 5376B]
  // total = 160768 B -> exactly 2 workgroups per 320KB WGP (4 waves = 4 SIMD32s)
  _Float16* WT   = (_Float16*)smem;
  _Float16* actA = (_Float16*)(smem + 139264 + wave * (2 * 16 * SA * 2));
  _Float16* actB = actA + 16 * SA;

  // ---------------- Phase 1: GRU over 128 timesteps ----------------
  _Float16* WihT  = WT;             // [32][SWG]  (rows 8..31 zero pad)
  _Float16* WhhT  = WT + 32 * SWG;  // [128][SWG], ends at byte 125440
  _Float16* rbuf  = (_Float16*)(smem + 125440 + wave * (16 * SRB * 2)); // r-gate f16
  for (int idx = tid; idx < 32 * SWG; idx += nth) WihT[idx] = (_Float16)0.f;
  __syncthreads();
  stageWT(gru_Wih, WihT, 384, 8,   SWG, tid, nth);
  stageWT(gru_Whh, WhhT, 384, 128, SWG, tid, nth);
  for (int idx = lane; idx < 16 * SA; idx += 32) {
    actA[idx] = (_Float16)0.f;    // h0 = 0 and zero padding
    actB[idx] = (_Float16)0.f;
  }
  // preload per-lane bias combos into registers (avoid global loads in hot loop)
  float brz[16], bni[8], bnh[8];
#pragma unroll
  for (int i = 0; i < 16; ++i) brz[i] = gru_bih[i * 16 + nn] + gru_bhh[i * 16 + nn];
#pragma unroll
  for (int ntb = 0; ntb < 8; ++ntb) {
    bni[ntb] = gru_bih[256 + ntb * 16 + nn];
    bnh[ntb] = gru_bhh[256 + ntb * 16 + nn];
  }
  __syncthreads();

  for (int t = 0; t < NT; ++t) {
    _Float16* cur = (t & 1) ? actB : actA;   // holds h_t; xm_t written below
    _Float16* nxt = (t & 1) ? actA : actB;   // receives h_{t+1}
    if (lane < 16) {
      int row = row0 + lane;
#pragma unroll
      for (int j = 0; j < 4; ++j) {
        cur[lane * SA + 128 + j] = (_Float16)x[(size_t)row * (NT * 4) + t * 4 + j];
        cur[lane * SA + 132 + j] = (_Float16)meta[row * 4 + j];
      }
    }
    // hoist the 5 A fragments (h: 4 K-chunks, xm: 1) -- shared by all 24 tiles
    v16h ah0 = ldsA(cur, SA, 0,   lane);
    v16h ah1 = ldsA(cur, SA, 32,  lane);
    v16h ah2 = ldsA(cur, SA, 64,  lane);
    v16h ah3 = ldsA(cur, SA, 96,  lane);
    v16h ax  = ldsA(cur, SA, 128, lane);

    // r gate (cols 0..127): sigmoid(xm@Wih_r + h@Whh_r + b) -> rbuf (f16)
#pragma unroll
    for (int nt = 0; nt < 8; ++nt) {
      int n0 = nt * 16;
      v8f c;
#pragma unroll
      for (int r = 0; r < 8; ++r) c[r] = brz[nt];
      c = wmma_f16(ah0, ldsB(WhhT, SWG, 0,  n0, lane), c);
      c = wmma_f16(ah1, ldsB(WhhT, SWG, 32, n0, lane), c);
      c = wmma_f16(ah2, ldsB(WhhT, SWG, 64, n0, lane), c);
      c = wmma_f16(ah3, ldsB(WhhT, SWG, 96, n0, lane), c);
      c = wmma_f16(ax,  ldsB(WihT, SWG, 0,  n0, lane), c);
#pragma unroll
      for (int r = 0; r < 8; ++r)
        rbuf[(mb + r) * SRB + n0 + nn] = (_Float16)sigmoid_f(c[r]);
    }
    // z gate (cols 128..255) -> fp32 registers
    v8f zgt[8];
#pragma unroll
    for (int nt = 0; nt < 8; ++nt) {
      int nw = 128 + nt * 16;
      v8f c;
#pragma unroll
      for (int r = 0; r < 8; ++r) c[r] = brz[8 + nt];
      c = wmma_f16(ah0, ldsB(WhhT, SWG, 0,  nw, lane), c);
      c = wmma_f16(ah1, ldsB(WhhT, SWG, 32, nw, lane), c);
      c = wmma_f16(ah2, ldsB(WhhT, SWG, 64, nw, lane), c);
      c = wmma_f16(ah3, ldsB(WhhT, SWG, 96, nw, lane), c);
      c = wmma_f16(ax,  ldsB(WihT, SWG, 0,  nw, lane), c);
#pragma unroll
      for (int r = 0; r < 8; ++r) zgt[nt][r] = sigmoid_f(c[r]);
    }
    // n gate (cols 256..383): ng = tanh(i_n + r*h_n); h' = (1-z)*ng + z*h
#pragma unroll
    for (int nt = 0; nt < 8; ++nt) {
      int n0 = nt * 16, nw = 256 + n0;
      v8f hn;
#pragma unroll
      for (int r = 0; r < 8; ++r) hn[r] = bnh[nt];
      hn = wmma_f16(ah0, ldsB(WhhT, SWG, 0,  nw, lane), hn);
      hn = wmma_f16(ah1, ldsB(WhhT, SWG, 32, nw, lane), hn);
      hn = wmma_f16(ah2, ldsB(WhhT, SWG, 64, nw, lane), hn);
      hn = wmma_f16(ah3, ldsB(WhhT, SWG, 96, nw, lane), hn);
      v8f in;
#pragma unroll
      for (int r = 0; r < 8; ++r) in[r] = bni[nt];
      in = wmma_f16(ax, ldsB(WihT, SWG, 0, nw, lane), in);
#pragma unroll
      for (int r = 0; r < 8; ++r) {
        float rv = (float)rbuf[(mb + r) * SRB + n0 + nn];
        float ng = fast_tanh(in[r] + rv * hn[r]);
        float ho = (float)cur[(mb + r) * SA + n0 + nn];
        float zg = zgt[nt][r];
        nxt[(mb + r) * SA + n0 + nn] = (_Float16)((1.f - zg) * ng + zg * ho);
      }
    }
  }
  // hN now in actA (t=127 wrote into actA)

  // ---------------- Phase 2: encoder + reparameterization ----------------
  __syncthreads();
  _Float16* W1T = WT;               // [128][SW1]
  _Float16* W2T = WT + 128 * SW1;   // [128][SW2]
  stageWT(enc_W1, W1T, 128, 128, SW1, tid, nth);
  stageWT(enc_W2, W2T, 256, 128, SW2, tid, nth);
  float eb1[8], ebm[8], ebs[8];
#pragma unroll
  for (int nt = 0; nt < 8; ++nt) {
    eb1[nt] = enc_b1[nt * 16 + nn];
    ebm[nt] = enc_b2[nt * 16 + nn];
    ebs[nt] = enc_b2[128 + nt * 16 + nn];
  }
  __syncthreads();

  {
    v16h a0 = ldsA(actA, SA, 0,  lane);
    v16h a1 = ldsA(actA, SA, 32, lane);
    v16h a2 = ldsA(actA, SA, 64, lane);
    v16h a3 = ldsA(actA, SA, 96, lane);
#pragma unroll
    for (int nt = 0; nt < 8; ++nt) {    // l1 = relu(hN @ W1^T + b1) -> actB
      int n0 = nt * 16;
      v8f c;
#pragma unroll
      for (int r = 0; r < 8; ++r) c[r] = eb1[nt];
      c = wmma_f16(a0, ldsB(W1T, SW1, 0,  n0, lane), c);
      c = wmma_f16(a1, ldsB(W1T, SW1, 32, n0, lane), c);
      c = wmma_f16(a2, ldsB(W1T, SW1, 64, n0, lane), c);
      c = wmma_f16(a3, ldsB(W1T, SW1, 96, n0, lane), c);
#pragma unroll
      for (int r = 0; r < 8; ++r) c[r] = fmaxf(c[r], 0.f);
      stD_tile(actB, SA, n0, c, lane);
    }
  }

  v8f zt[8];                          // fp32 latent state, D layout, lives in VGPRs
  {
    v16h a0 = ldsA(actB, SA, 0,  lane);
    v16h a1 = ldsA(actB, SA, 32, lane);
    v16h a2 = ldsA(actB, SA, 64, lane);
    v16h a3 = ldsA(actB, SA, 96, lane);
#pragma unroll
    for (int nt = 0; nt < 8; ++nt) {  // enc2: mean (cols 0..127), std (128..255)
      int n0 = nt * 16;
      v8f cm, cs;
#pragma unroll
      for (int r = 0; r < 8; ++r) { cm[r] = ebm[nt]; cs[r] = ebs[nt]; }
      cm = wmma_f16(a0, ldsB(W2T, SW2, 0,  n0, lane), cm);
      cm = wmma_f16(a1, ldsB(W2T, SW2, 32, n0, lane), cm);
      cm = wmma_f16(a2, ldsB(W2T, SW2, 64, n0, lane), cm);
      cm = wmma_f16(a3, ldsB(W2T, SW2, 96, n0, lane), cm);
      cs = wmma_f16(a0, ldsB(W2T, SW2, 0,  128 + n0, lane), cs);
      cs = wmma_f16(a1, ldsB(W2T, SW2, 32, 128 + n0, lane), cs);
      cs = wmma_f16(a2, ldsB(W2T, SW2, 64, 128 + n0, lane), cs);
      cs = wmma_f16(a3, ldsB(W2T, SW2, 96, 128 + n0, lane), cs);
#pragma unroll
      for (int r = 0; r < 8; ++r) {
        float ev = eps[(size_t)(row0 + mb + r) * HD + nt * 16 + nn];
        zt[nt][r] = cm[r] + ev * cs[r];   // z0 = mean + eps*std
      }
    }
  }

  // ---------------- Phase 3: ODE, 127 steps x 4 RK4 substeps ----------------
  __syncthreads();
  _Float16* O1T = WT;
  _Float16* O2T = WT + 128 * SWO;
  _Float16* O3T = WT + 2 * 128 * SWO;
  _Float16* O4T = WT + 3 * 128 * SWO;
  stageWT(ode_W1, O1T, 128, 128, SWO, tid, nth);
  stageWT(ode_W2, O2T, 128, 128, SWO, tid, nth);
  stageWT(ode_W3, O3T, 128, 128, SWO, tid, nth);
  stageWT(ode_W4, O4T, 128, 128, SWO, tid, nth);
  float ob1[8], ob2[8], ob3[8], ob4[8];
#pragma unroll
  for (int nt = 0; nt < 8; ++nt) {
    ob1[nt] = ode_b1[nt * 16 + nn];
    ob2[nt] = ode_b2[nt * 16 + nn];
    ob3[nt] = ode_b3[nt * 16 + nn];
    ob4[nt] = ode_b4[nt * 16 + nn];
  }
  __syncthreads();

  _Float16* b0 = actA;
  _Float16* b1 = actB;

  for (int t = 0; t < NT - 1; ++t) {
    float hv[8], dv[8];
#pragma unroll
    for (int r = 0; r < 8; ++r) {
      size_t base = (size_t)(row0 + mb + r) * (NT * 4) + (size_t)t * 4;
      float t0 = x[base];
      float t1 = x[base + 4];
      dv[r] = x[base + 2];              // dose
      hv[r] = (t1 - t0) * 0.25f;        // dt / NSTEPS
    }
#pragma unroll
    for (int nt = 0; nt < 8; ++nt)
#pragma unroll
      for (int r = 0; r < 8; ++r) zt[nt][r] += dv[r];   // z += dose[:,None]

    for (int sub = 0; sub < 4; ++sub) { // NSTEPS rk4 substeps
#pragma unroll
      for (int nt = 0; nt < 8; ++nt) stD_tile(b0, SA, nt * 16, zt[nt], lane);
      v8f acc[8];
#pragma unroll
      for (int nt = 0; nt < 8; ++nt)
#pragma unroll
        for (int r = 0; r < 8; ++r) acc[nt][r] = 0.f;

      for (int s = 0; s < 4; ++s) {     // rk4 stages k1..k4
        const float wa = (s == 1 || s == 2) ? 2.f : 1.f;
        const float cn = (s < 2) ? 0.5f : 1.f;
        layer_selu(b0, b1, O1T, ob1, lane);
        layer_selu(b1, b0, O2T, ob2, lane);
        layer_selu(b0, b1, O3T, ob3, lane);
        // layer 4 (linear) fused with RK4 stage combine; A frags hoisted
        v16h a0 = ldsA(b1, SA, 0,  lane);
        v16h a1 = ldsA(b1, SA, 32, lane);
        v16h a2 = ldsA(b1, SA, 64, lane);
        v16h a3 = ldsA(b1, SA, 96, lane);
#pragma unroll
        for (int nt = 0; nt < 8; ++nt) {
          int n0 = nt * 16;
          v8f k;
#pragma unroll
          for (int r = 0; r < 8; ++r) k[r] = ob4[nt];
          k = wmma_f16(a0, ldsB(O4T, SWO, 0,  n0, lane), k);
          k = wmma_f16(a1, ldsB(O4T, SWO, 32, n0, lane), k);
          k = wmma_f16(a2, ldsB(O4T, SWO, 64, n0, lane), k);
          k = wmma_f16(a3, ldsB(O4T, SWO, 96, n0, lane), k);
          if (s < 3) {
            v8f tmp;
#pragma unroll
            for (int r = 0; r < 8; ++r) {
              acc[nt][r] += wa * k[r];
              tmp[r] = zt[nt][r] + cn * hv[r] * k[r];   // next stage input
            }
            stD_tile(b0, SA, n0, tmp, lane);
          } else {
#pragma unroll
            for (int r = 0; r < 8; ++r) acc[nt][r] += k[r];
          }
        }
      }
#pragma unroll
      for (int nt = 0; nt < 8; ++nt)
#pragma unroll
        for (int r = 0; r < 8; ++r)
          zt[nt][r] += (hv[r] * (1.f / 6.f)) * acc[nt][r];
    }
  }

  // ---------------- Phase 4: final FC (H+META -> 1), shuffle reduction ----------
  float p[8];
#pragma unroll
  for (int r = 0; r < 8; ++r) p[r] = 0.f;
#pragma unroll
  for (int nt = 0; nt < 8; ++nt) {
    float w = fc_W[nt * 16 + nn];
#pragma unroll
    for (int r = 0; r < 8; ++r) p[r] += zt[nt][r] * w;
  }
#pragma unroll
  for (int r = 0; r < 8; ++r) {
#pragma unroll
    for (int m = 1; m < 16; m <<= 1)
      p[r] += __shfl_xor(p[r], m, 32);  // reduce over n within each 16-lane half
  }
  if (nn == 0) {
    float fb = fc_b[0];
#pragma unroll
    for (int r = 0; r < 8; ++r) {
      int row = row0 + mb + r;
      float v = p[r] + fb;
#pragma unroll
      for (int j = 0; j < 4; ++j) v += meta[row * 4 + j] * fc_W[128 + j];
      out[row] = v;
    }
  }
}

extern "C" void kernel_launch(void* const* d_in, const int* in_sizes, int n_in,
                              void* d_out, int out_size, void* d_ws, size_t ws_size,
                              hipStream_t stream) {
  (void)in_sizes; (void)n_in; (void)out_size; (void)d_ws; (void)ws_size;
  const float* x       = (const float*)d_in[0];
  const float* meta    = (const float*)d_in[1];
  const float* eps     = (const float*)d_in[2];
  const float* gru_Wih = (const float*)d_in[3];
  const float* gru_Whh = (const float*)d_in[4];
  const float* gru_bih = (const float*)d_in[5];
  const float* gru_bhh = (const float*)d_in[6];
  const float* enc_W1  = (const float*)d_in[7];
  const float* enc_b1  = (const float*)d_in[8];
  const float* enc_W2  = (const float*)d_in[9];
  const float* enc_b2  = (const float*)d_in[10];
  const float* ode_W1  = (const float*)d_in[11];
  const float* ode_b1  = (const float*)d_in[12];
  const float* ode_W2  = (const float*)d_in[13];
  const float* ode_b2  = (const float*)d_in[14];
  const float* ode_W3  = (const float*)d_in[15];
  const float* ode_b3  = (const float*)d_in[16];
  const float* ode_W4  = (const float*)d_in[17];
  const float* ode_b4  = (const float*)d_in[18];
  const float* fc_W    = (const float*)d_in[19];
  const float* fc_b    = (const float*)d_in[20];
  float* outp = (float*)d_out;

  const size_t shmem = 160768; // weights 139264 (rbuf16 in tail) + act 21504
  hipFuncSetAttribute(reinterpret_cast<const void*>(&node_kernel),
                      hipFuncAttributeMaxDynamicSharedMemorySize, (int)shmem);
  node_kernel<<<dim3(NB / 32), dim3(64), shmem, stream>>>(
      x, meta, eps, gru_Wih, gru_Whh, gru_bih, gru_bhh,
      enc_W1, enc_b1, enc_W2, enc_b2,
      ode_W1, ode_b1, ode_W2, ode_b2, ode_W3, ode_b3, ode_W4, ode_b4,
      fc_W, fc_b, outp);
}
